// NeighborhoodSelfAttentionBlock_20839181320568
// MI455X (gfx1250) — matrix operand order repaired and checked by reading, not compiled, not run on hardware
//
#include <hip/hip_runtime.h>

// ---------------------------------------------------------------------------
// Neighborhood self-attention block for MI455X (gfx1250, wave32, WMMA).
// GEMMs (qkv proj, out proj) run on v_wmma_f32_16x16x32_bf16 with f32 accum.
// Activation tiles are staged into LDS with GLOBAL_LOAD_ASYNC_TO_LDS_B128
// (ASYNCcnt / s_wait_asynccnt). B-fragment loads are software-pipelined so
// the WMMA chain never stalls on loadcnt==0. Everything else (rmsnorm,
// qk-norm+RoPE, 7x7 softmax-attention) is VALU. Working set is L2-resident.
// ---------------------------------------------------------------------------

#define D_MODEL 512
#define D_HEAD  64
#define N_HEADS 8
#define KSZ     7
#define BB      2
#define HH      48
#define WW      48
#define NPIX    (HH * WW)        // 2304
#define TOTPIX  (BB * NPIX)      // 4608
#define COND_D  768
#define EPSF    1e-6f

typedef __attribute__((ext_vector_type(16))) __bf16       v16bf;
typedef __attribute__((ext_vector_type(8)))  float        v8f;
typedef __attribute__((ext_vector_type(4)))  unsigned int u32x4;
typedef __attribute__((ext_vector_type(4)))  int          i32x4;

union Frag {
  v16bf v;
  u32x4 q[2];
};

#if __has_builtin(__builtin_amdgcn_global_load_async_to_lds_b128)
#define HAVE_ASYNC_LDS 1
#else
#define HAVE_ASYNC_LDS 0
#endif

__device__ __forceinline__ void wait_async_lds() {
#if HAVE_ASYNC_LDS
#if __has_builtin(__builtin_amdgcn_s_wait_asynccnt)
  __builtin_amdgcn_s_wait_asynccnt(0);
#else
  asm volatile("s_wait_asynccnt 0x0" ::: "memory");
#endif
#endif
}

// Stage a contiguous 16x512 f32 tile (32KB = 2048 x 16B chunks) from global
// into LDS. 256 threads x 8 chunks. Caller must __syncthreads() afterwards.
__device__ __forceinline__ void stage_tile(const float* __restrict__ gsrc,
                                           float* ldst, int tid) {
#if HAVE_ASYNC_LDS
  for (int i = 0; i < 8; ++i) {
    const int chunk = i * 256 + tid;
    __builtin_amdgcn_global_load_async_to_lds_b128(
        (__attribute__((address_space(1))) i32x4*)(gsrc + chunk * 4),
        (__attribute__((address_space(3))) i32x4*)(ldst + chunk * 4),
        0, 0);
  }
  wait_async_lds();
#else
  for (int i = 0; i < 8; ++i) {
    const int chunk = i * 256 + tid;
    *(float4*)(ldst + chunk * 4) = *(const float4*)(gsrc + chunk * 4);
  }
#endif
}

__device__ __forceinline__ unsigned short f2bf(float f) {
  unsigned int u = __float_as_uint(f);
  u += 0x7FFFu + ((u >> 16) & 1u);   // round-to-nearest-even
  return (unsigned short)(u >> 16);
}

__device__ __forceinline__ void load_bfrag(Frag& f, const unsigned short* p) {
  f.q[0] = *(const u32x4*)(p);
  f.q[1] = *(const u32x4*)(p + 8);
}

// ---------------------------------------------------------------------------
// K0: ada = cond @ norm_w.T + 1   (2 x 512 outputs, K=768)
// ---------------------------------------------------------------------------
__global__ __launch_bounds__(256) void k_ada(const float* __restrict__ cond,
                                             const float* __restrict__ nw,
                                             float* __restrict__ adaP1) {
  int t = blockIdx.x * 256 + threadIdx.x;      // 0..1023
  if (t >= BB * D_MODEL) return;
  int b = t >> 9;
  int c = t & (D_MODEL - 1);
  const float* cp = cond + (size_t)b * COND_D;
  const float* wp = nw + (size_t)c * COND_D;
  float s = 0.f;
  for (int i = 0; i < COND_D; ++i) s += cp[i] * wp[i];
  adaP1[t] = s + 1.0f;
}

// ---------------------------------------------------------------------------
// K0b: convert qkv_w (1536x512) and out_w (512x512) f32 -> bf16 (row major)
// ---------------------------------------------------------------------------
__global__ __launch_bounds__(256) void k_cvt(const float* __restrict__ qkvw,
                                             const float* __restrict__ outw,
                                             unsigned short* __restrict__ qbf,
                                             unsigned short* __restrict__ obf) {
  int t = blockIdx.x * 256 + threadIdx.x;
  const int NQ = 3 * D_MODEL * D_MODEL;        // 786432
  const int NO = D_MODEL * D_MODEL;            // 262144
  if (t < NQ) {
    qbf[t] = f2bf(qkvw[t]);
  } else if (t < NQ + NO) {
    int u = t - NQ;
    obf[u] = f2bf(outw[u]);
  }
}

// ---------------------------------------------------------------------------
// K1: fused ada-RMSNorm + QKV projection via WMMA bf16.
// One block = 16-pixel M tile, 8 waves, each wave owns 12 N tiles (N=1536).
// x tile is async-staged into LDS once; both the sum-of-squares pass and the
// bf16 conversion read LDS. q/k/v written as [b][head][h*w][64] f32.
// ---------------------------------------------------------------------------
__global__ __launch_bounds__(256) void k_qkv(const float* __restrict__ x,
                                             const float* __restrict__ adaP1,
                                             const unsigned short* __restrict__ wbf,
                                             float* __restrict__ q,
                                             float* __restrict__ k,
                                             float* __restrict__ v) {
  __shared__ float          sX[16 * D_MODEL];   // 32KB staged f32 tile
  __shared__ unsigned short sXN[16 * D_MODEL];  // 16KB bf16 A tile
  __shared__ float sRed[16][17];
  __shared__ float sInv[16];

  const int tid = threadIdx.x;
  const int m0  = blockIdx.x * 16;              // tile never crosses batch (2304 % 16 == 0)
  const int b   = m0 / NPIX;
  const int hw0 = m0 % NPIX;

  // ---- phase 0: async stage x tile into LDS ----
  stage_tile(x + (size_t)m0 * D_MODEL, sX, tid);
  __syncthreads();

  // ---- phase 1: sum of squares per pixel (from LDS) ----
  const int row = tid >> 4;                     // pixel within tile
  const int sub = tid & 15;
  const float* xr = sX + row * D_MODEL;
  float ss = 0.f;
  for (int i = 0; i < 32; ++i) {
    float t = xr[sub + i * 16];
    ss += t * t;
  }
  sRed[row][sub] = ss;
  __syncthreads();
  if (sub == 0) {
    float s = 0.f;
    for (int i = 0; i < 16; ++i) s += sRed[row][i];
    sInv[row] = rsqrtf(s * (1.0f / D_MODEL) + EPSF);
  }
  __syncthreads();

  // ---- phase 2: normalized bf16 tile into LDS ----
  const float* adab = adaP1 + (size_t)b * D_MODEL;
  const float inv = sInv[row];
  for (int i = 0; i < 32; ++i) {
    int c = sub + i * 16;
    sXN[row * D_MODEL + c] = f2bf(xr[c] * adab[c] * inv);
  }
  __syncthreads();

  // ---- phase 3: WMMA GEMM, K = 512 (16 steps of 32), B double-buffered ----
  const int lane = tid & 31;
  const int wv   = tid >> 5;
  const int mrow = lane & 15;
  const int hi   = lane >> 4;

  for (int nt = 0; nt < 12; ++nt) {
    const int n0 = (wv * 12 + nt) * 16;
    const unsigned short* wrow = wbf + (size_t)(n0 + mrow) * D_MODEL;
    v8f acc = {};
    Frag fb[2];
    load_bfrag(fb[0], wrow + hi * 16);          // prologue: kk = 0
    for (int kk = 0; kk < 16; ++kk) {
      const int cur = kk & 1;
      if (kk < 15) {                            // prefetch kk+1 before use
        load_bfrag(fb[cur ^ 1], wrow + (kk + 1) * 32 + hi * 16);
      }
      Frag fa;
      const unsigned short* ar = sXN + mrow * D_MODEL + kk * 32;
      fa.q[0] = *(const u32x4*)(ar + hi * 16);
      fa.q[1] = *(const u32x4*)(ar + hi * 16 + 8);
      acc = __builtin_amdgcn_wmma_f32_16x16x32_bf16(
          false, fa.v, false, fb[cur].v, (short)0, acc, false, false);
    }
    // ---- epilogue: scatter to q/k/v [b][head][hw][e] ----
    const int t3   = n0 / D_MODEL;              // 0=q 1=k 2=v
    const int j0   = n0 % D_MODEL;
    const int head = j0 / D_HEAD;
    const int e0   = j0 % D_HEAD;               // tile lies inside one head
    float* dst = (t3 == 0) ? q : (t3 == 1) ? k : v;
    for (int i = 0; i < 8; ++i) {
      const int m = i + 8 * hi;
      size_t off = ((size_t)(b * N_HEADS + head) * NPIX + (hw0 + m)) * D_HEAD
                   + (e0 + mrow);
      dst[off] = acc[i];
    }
  }
}

// ---------------------------------------------------------------------------
// K2: per (q|k, b, head, pixel): L2-normalize * sqrt(scale), then RoPE.
// ---------------------------------------------------------------------------
__global__ __launch_bounds__(256) void k_qknorm_rope(float* __restrict__ q,
                                                     float* __restrict__ k,
                                                     const float* __restrict__ pos,
                                                     const float* __restrict__ scale) {
  const int t = blockIdx.x * 256 + threadIdx.x;     // 0 .. 2*2*8*2304-1
  const int NITEMS = BB * N_HEADS * NPIX;           // 36864
  if (t >= 2 * NITEMS) return;
  const int which = t / NITEMS;                     // 0=q 1=k
  const int item  = t % NITEMS;
  const int bn    = item / NPIX;
  const int hw    = item % NPIX;
  const int b     = bn / N_HEADS;
  const int head  = bn % N_HEADS;

  float* base = (which ? k : q) + (size_t)item * D_HEAD;

  float vb[D_HEAD];
  const float4* p4 = (const float4*)base;
  for (int i = 0; i < 16; ++i) {
    float4 r = p4[i];
    vb[4 * i + 0] = r.x; vb[4 * i + 1] = r.y;
    vb[4 * i + 2] = r.z; vb[4 * i + 3] = r.w;
  }
  float ssq = 0.f;
  for (int i = 0; i < D_HEAD; ++i) ssq += vb[i] * vb[i];
  const float fac = sqrtf(scale[head]) * rsqrtf(ssq + EPSF);
  for (int i = 0; i < D_HEAD; ++i) vb[i] *= fac;

  // RoPE: freqs[head][i] = exp(log(pi) + (i*8+head) * log(10)/64), i=0..7
  const float py = pos[((size_t)b * NPIX + hw) * 2 + 0];
  const float px = pos[((size_t)b * NPIX + hw) * 2 + 1];
  const float lmin = 1.14472988584940017f;          // log(pi)
  const float dlt  = 2.30258509299404568f / 64.0f;  // log(10)/64
  for (int i = 0; i < 16; ++i) {
    const int fi = (i < 8) ? i : (i - 8);
    const float f  = __expf(lmin + (float)(fi * 8 + head) * dlt);
    const float th = ((i < 8) ? py : px) * f;
    float c, s;
    __sincosf(th, &s, &c);
    const float x1 = vb[i];
    const float x2 = vb[i + 16];
    vb[i]      = x1 * c - x2 * s;
    vb[i + 16] = x2 * c + x1 * s;
  }

  float4* o4 = (float4*)base;
  for (int i = 0; i < 16; ++i) {
    float4 r;
    r.x = vb[4 * i + 0]; r.y = vb[4 * i + 1];
    r.z = vb[4 * i + 2]; r.w = vb[4 * i + 3];
    o4[i] = r;
  }
}

// ---------------------------------------------------------------------------
// K3: 7x7 neighborhood attention. One wave per (b, head, pixel).
// Lanes split the 49 neighbors for scores; shfl softmax; per-lane 2 channels
// of the 64-wide V accumulation. Writes attn [pixel][head*64+e] f32.
// ---------------------------------------------------------------------------
__global__ __launch_bounds__(256) void k_attn(const float* __restrict__ q,
                                              const float* __restrict__ k,
                                              const float* __restrict__ v,
                                              float* __restrict__ attn) {
  __shared__ float qs[8][D_HEAD];
  __shared__ float sc[8][64];

  const int wv   = threadIdx.x >> 5;
  const int lane = threadIdx.x & 31;
  const int item = blockIdx.x * 8 + wv;          // < 36864
  const int bn   = item / NPIX;
  const int hw   = item % NPIX;
  const int b    = bn / N_HEADS;
  const int head = bn % N_HEADS;
  const int h    = hw / WW;
  const int w    = hw % WW;
  int rs = h - KSZ / 2; rs = rs < 0 ? 0 : (rs > HH - KSZ ? HH - KSZ : rs);
  int cs = w - KSZ / 2; cs = cs < 0 ? 0 : (cs > WW - KSZ ? WW - KSZ : cs);

  const float* qp = q + (size_t)item * D_HEAD;
  qs[wv][lane]      = qp[lane];
  qs[wv][lane + 32] = qp[lane + 32];
  __syncthreads();

  const float* kbase = k + (size_t)bn * NPIX * D_HEAD;
  const float* vbase = v + (size_t)bn * NPIX * D_HEAD;

  float d0 = -1e30f, d1 = -1e30f;
  {
    const int j = lane;                           // always < 49
    const int r = rs + j / KSZ, c = cs + j % KSZ;
    const float4* kp4 = (const float4*)(kbase + (size_t)(r * WW + c) * D_HEAD);
    float acc = 0.f;
    for (int e = 0; e < 16; ++e) {
      float4 kv = kp4[e];
      acc += qs[wv][4 * e + 0] * kv.x + qs[wv][4 * e + 1] * kv.y
           + qs[wv][4 * e + 2] * kv.z + qs[wv][4 * e + 3] * kv.w;
    }
    d0 = acc;
  }
  if (lane < KSZ * KSZ - 32) {                    // lanes 0..16 cover j=32..48
    const int j = lane + 32;
    const int r = rs + j / KSZ, c = cs + j % KSZ;
    const float4* kp4 = (const float4*)(kbase + (size_t)(r * WW + c) * D_HEAD);
    float acc = 0.f;
    for (int e = 0; e < 16; ++e) {
      float4 kv = kp4[e];
      acc += qs[wv][4 * e + 0] * kv.x + qs[wv][4 * e + 1] * kv.y
           + qs[wv][4 * e + 2] * kv.z + qs[wv][4 * e + 3] * kv.w;
    }
    d1 = acc;
  }

  float m = fmaxf(d0, d1);
  for (int off = 16; off; off >>= 1) m = fmaxf(m, __shfl_xor(m, off, 32));

  const float p0 = __expf(d0 - m);
  const float p1 = (lane < KSZ * KSZ - 32) ? __expf(d1 - m) : 0.f;
  float s = p0 + p1;
  for (int off = 16; off; off >>= 1) s += __shfl_xor(s, off, 32);

  sc[wv][lane] = p0;
  if (lane < KSZ * KSZ - 32) sc[wv][lane + 32] = p1;
  __syncthreads();

  const float invs = 1.0f / s;
  const int e0 = lane * 2;
  float a0 = 0.f, a1 = 0.f;
  for (int j = 0; j < KSZ * KSZ; ++j) {
    const float p = sc[wv][j];
    const int r = rs + j / KSZ, c = cs + j % KSZ;
    const float* vp = vbase + (size_t)(r * WW + c) * D_HEAD;
    a0 += p * vp[e0];
    a1 += p * vp[e0 + 1];
  }
  float* op = attn + ((size_t)(b * NPIX + hw) * D_MODEL + head * D_HEAD + e0);
  op[0] = a0 * invs;
  op[1] = a1 * invs;
}

// ---------------------------------------------------------------------------
// K4: out = attn @ out_w.T + skip(x) via WMMA bf16. Same tiling as K1 (N=512).
// attn tile async-staged into LDS, converted to bf16 in LDS.
// ---------------------------------------------------------------------------
__global__ __launch_bounds__(256) void k_outproj(const float* __restrict__ attn,
                                                 const unsigned short* __restrict__ wbf,
                                                 const float* __restrict__ x,
                                                 float* __restrict__ out) {
  __shared__ float          sA[16 * D_MODEL];   // 32KB staged f32 tile
  __shared__ unsigned short sAN[16 * D_MODEL];  // 16KB bf16 A tile

  const int tid = threadIdx.x;
  const int m0  = blockIdx.x * 16;

  stage_tile(attn + (size_t)m0 * D_MODEL, sA, tid);
  __syncthreads();

  const int row = tid >> 4;
  const int sub = tid & 15;
  const float* ar0 = sA + row * D_MODEL;
  for (int i = 0; i < 32; ++i) {
    int c = sub + i * 16;
    sAN[row * D_MODEL + c] = f2bf(ar0[c]);
  }
  __syncthreads();

  const int lane = tid & 31;
  const int wv   = tid >> 5;
  const int mrow = lane & 15;
  const int hi   = lane >> 4;

  for (int nt = 0; nt < 4; ++nt) {              // 32 N-tiles / 8 waves
    const int n0 = (wv * 4 + nt) * 16;
    const unsigned short* wrow = wbf + (size_t)(n0 + mrow) * D_MODEL;
    v8f acc = {};
    Frag fb[2];
    load_bfrag(fb[0], wrow + hi * 16);          // prologue: kk = 0
    for (int kk = 0; kk < 16; ++kk) {
      const int cur = kk & 1;
      if (kk < 15) {                            // prefetch kk+1 before use
        load_bfrag(fb[cur ^ 1], wrow + (kk + 1) * 32 + hi * 16);
      }
      Frag fa;
      const unsigned short* ar = sAN + mrow * D_MODEL + kk * 32;
      fa.q[0] = *(const u32x4*)(ar + hi * 16);
      fa.q[1] = *(const u32x4*)(ar + hi * 16 + 8);
      acc = __builtin_amdgcn_wmma_f32_16x16x32_bf16(
          false, fa.v, false, fb[cur].v, (short)0, acc, false, false);
    }
    for (int i = 0; i < 8; ++i) {
      const int m = i + 8 * hi;
      const size_t off = (size_t)(m0 + m) * D_MODEL + (n0 + mrow);
      out[off] = acc[i] + x[off];               // skip connection
    }
  }
}

// ---------------------------------------------------------------------------
// Host side
// ---------------------------------------------------------------------------
extern "C" void kernel_launch(void* const* d_in, const int* in_sizes, int n_in,
                              void* d_out, int out_size, void* d_ws, size_t ws_size,
                              hipStream_t stream) {
  (void)in_sizes; (void)n_in; (void)out_size; (void)ws_size;

  const float* x      = (const float*)d_in[0];
  const float* pos    = (const float*)d_in[1];
  const float* cond   = (const float*)d_in[2];
  const float* norm_w = (const float*)d_in[3];
  const float* qkv_w  = (const float*)d_in[4];
  const float* scale  = (const float*)d_in[5];
  const float* out_w  = (const float*)d_in[6];
  float* out = (float*)d_out;

  // workspace layout (~38 MB needed)
  char* ws = (char*)d_ws;
  const size_t OFF_ADA = 0;                                        // 4 KB
  const size_t OFF_QW  = 4096;                                     // 1.5 MB bf16
  const size_t OFF_OW  = OFF_QW + (size_t)3 * D_MODEL * D_MODEL * 2;
  const size_t OFF_Q   = OFF_OW + (size_t)D_MODEL * D_MODEL * 2;
  const size_t SZ_T    = (size_t)TOTPIX * D_MODEL * sizeof(float); // 9.44 MB
  const size_t OFF_K   = OFF_Q + SZ_T;
  const size_t OFF_V   = OFF_K + SZ_T;
  const size_t OFF_A   = OFF_V + SZ_T;

  float*          adaP1   = (float*)(ws + OFF_ADA);
  unsigned short* qkvw_bf = (unsigned short*)(ws + OFF_QW);
  unsigned short* outw_bf = (unsigned short*)(ws + OFF_OW);
  float*          q       = (float*)(ws + OFF_Q);
  float*          k       = (float*)(ws + OFF_K);
  float*          v       = (float*)(ws + OFF_V);
  float*          attn    = (float*)(ws + OFF_A);

  k_ada<<<dim3(4), dim3(256), 0, stream>>>(cond, norm_w, adaP1);
  k_cvt<<<dim3(4096), dim3(256), 0, stream>>>(qkv_w, out_w, qkvw_bf, outw_bf);
  k_qkv<<<dim3(TOTPIX / 16), dim3(256), 0, stream>>>(x, adaP1, qkvw_bf, q, k, v);
  k_qknorm_rope<<<dim3(288), dim3(256), 0, stream>>>(q, k, pos, scale);
  k_attn<<<dim3(BB * N_HEADS * NPIX / 8), dim3(256), 0, stream>>>(q, k, v, attn);
  k_outproj<<<dim3(TOTPIX / 16), dim3(256), 0, stream>>>(attn, outw_bf, x, out);
}
